// CrossMamba_55490977464368
// MI455X (gfx1250) — compile-verified, matrix operand order checked by reading, not run on hardware
//
#include <hip/hip_runtime.h>
#include <hip/hip_bf16.h>
#include <stdint.h>

// Problem constants (match reference)
#define cB   2
#define cL   1024
#define cDM  1024
#define cDI  2048
#define cN   16
#define cR   64
#define cKC  4            // conv kernel width
#define cMB  (cB * cL)    // 2048 tokens
#define cSPLITK 8         // split-K factor for the narrow (N=96) GEMM
#define EPSV 1e-5f

typedef __attribute__((ext_vector_type(16))) __bf16 v16bf;
typedef __attribute__((ext_vector_type(8)))  float  v8f;

union FragBF { uint4 u4[2]; v16bf v; };   // 32 bytes: 16 bf16 values

__device__ __forceinline__ unsigned short f2bf(float x) {
    // round-to-nearest-even f32 -> bf16
    unsigned int u = __float_as_uint(x);
    unsigned int r = u + 0x7FFFu + ((u >> 16) & 1u);
    return (unsigned short)(r >> 16);
}
__device__ __forceinline__ float sigmoidf_(float x) { return 1.f / (1.f + __expf(-x)); }
__device__ __forceinline__ float softplusf_(float x) {
    return (x > 20.f) ? x : log1pf(__expf(x));
}

// ---------------- pack / layout kernels ----------------

__global__ void cm_pack(const float* __restrict__ in, unsigned short* __restrict__ out, int n) {
    int i = blockIdx.x * blockDim.x + threadIdx.x;
    if (i < n) out[i] = f2bf(in[i]);
}

// b (B,L,DM) f32 -> flipped-along-L bf16
__global__ void cm_pack_flip(const float* __restrict__ in, unsigned short* __restrict__ out) {
    int i = blockIdx.x * blockDim.x + threadIdx.x;
    if (i >= cMB * cDM) return;
    int d  = i % cDM;
    int bl = i / cDM;
    int l  = bl % cL;
    int b  = bl / cL;
    out[((size_t)(b * cL + (cL - 1 - l))) * cDM + d] = f2bf(in[i]);
}

// ---------------- WMMA GEMM: C(MxN) = A(MxK) * W(NxK)^T, 32x32 tile / wave ----------
// KS: row stride of A/W in elements (compile-time). KL: k-loop length (compile-time;
//     KL < KS only for split-K launches, where blockIdx.z selects the k-chunk).
// MODE 0: plain f32 store.
// MODE 1: C = softplus(acc + 2*bias[n])     (delta projection)
// MODE 3: split-K partial store to C + blockIdx.z*M*N (deterministic reduction later)
template <int MODE>
__device__ __forceinline__ void cm_store_tile(float* __restrict__ C,
                                              const float* __restrict__ bias,
                                              int N, int m, int n, int l15, int hi8,
                                              v8f acc) {
    const int col = n + l15;
    float bz = 0.f;
    if (MODE == 1) bz = 2.0f * bias[col];
#pragma unroll
    for (int r = 0; r < 8; ++r) {
        float v = acc[r];
        if (MODE == 1) v = softplusf_(v + bz);
        C[(size_t)(m + hi8 + r) * N + col] = v;
    }
}

template <int MODE, int KS, int KL>
__global__ void __launch_bounds__(256)
cm_gemm(const unsigned short* __restrict__ A,
        const unsigned short* __restrict__ W,
        float* __restrict__ C,
        int M, int N,
        const float* __restrict__ bias) {
    const int lane = threadIdx.x;            // 0..31
    const int wave = threadIdx.y;            // 0..7
    const int n0 = (blockIdx.x * 8 + wave) * 32;   // 2 adjacent n-tiles / wave
    const int m0 = blockIdx.y * 32;                // 2 adjacent m-tiles / wave
    if (n0 >= N) return;

    const int kBeg = (MODE == 3) ? (blockIdx.z * KL) : 0;   // compile-time trip count KL/32

    const int l15 = lane & 15;
    const int hi8 = (lane >> 4) * 8;         // K sub-offset per §7.12.2 A-layout
    const unsigned short* a0 = A + (size_t)(m0 + l15) * KS + kBeg;
    const unsigned short* a1 = A + (size_t)(m0 + 16 + l15) * KS + kBeg;
    const unsigned short* w0 = W + (size_t)(n0 + l15) * KS + kBeg;
    const unsigned short* w1 = W + (size_t)(n0 + 16 + l15) * KS + kBeg;

    v8f acc00 = {}, acc01 = {}, acc10 = {}, acc11 = {};
#pragma unroll 2
    for (int kc = 0; kc < KL; kc += 32) {
        FragBF fa0, fa1, fb0, fb1;
        fa0.u4[0] = *(const uint4*)(a0 + kc + hi8);
        fa0.u4[1] = *(const uint4*)(a0 + kc + 16 + hi8);
        fa1.u4[0] = *(const uint4*)(a1 + kc + hi8);
        fa1.u4[1] = *(const uint4*)(a1 + kc + 16 + hi8);
        fb0.u4[0] = *(const uint4*)(w0 + kc + hi8);
        fb0.u4[1] = *(const uint4*)(w0 + kc + 16 + hi8);
        fb1.u4[0] = *(const uint4*)(w1 + kc + hi8);
        fb1.u4[1] = *(const uint4*)(w1 + kc + 16 + hi8);
        acc00 = __builtin_amdgcn_wmma_f32_16x16x32_bf16(false, fa0.v, false, fb0.v,
                                                        (short)0, acc00, false, false);
        acc01 = __builtin_amdgcn_wmma_f32_16x16x32_bf16(false, fa0.v, false, fb1.v,
                                                        (short)0, acc01, false, false);
        acc10 = __builtin_amdgcn_wmma_f32_16x16x32_bf16(false, fa1.v, false, fb0.v,
                                                        (short)0, acc10, false, false);
        acc11 = __builtin_amdgcn_wmma_f32_16x16x32_bf16(false, fa1.v, false, fb1.v,
                                                        (short)0, acc11, false, false);
    }

    float* Cd = (MODE == 3) ? (C + (size_t)blockIdx.z * ((size_t)M * N)) : C;
    constexpr int SM = (MODE == 3) ? 0 : MODE;   // partial stores use plain mode
    cm_store_tile<SM>(Cd, bias, N, m0,      n0,      l15, hi8, acc00);
    cm_store_tile<SM>(Cd, bias, N, m0,      n0 + 16, l15, hi8, acc01);
    cm_store_tile<SM>(Cd, bias, N, m0 + 16, n0,      l15, hi8, acc10);
    cm_store_tile<SM>(Cd, bias, N, m0 + 16, n0 + 16, l15, hi8, acc11);
}

// deterministic split-K reduction (fixed order) + fused dt bf16 extraction
__global__ void cm_reduce_dbl(const float* __restrict__ P, float* __restrict__ C,
                              unsigned short* __restrict__ Cbf) {
    int i = blockIdx.x * blockDim.x + threadIdx.x;   // over MB*96
    if (i >= cMB * 96) return;
    float s = 0.f;
#pragma unroll
    for (int z = 0; z < cSPLITK; ++z) s += P[(size_t)z * (cMB * 96) + i];
    C[i] = s;
    int col = i % 96;
    if (col < cR) Cbf[(size_t)(i / 96) * cR + col] = f2bf(s);
}

// ---------------- causal depthwise conv (K=4) + SiLU ----------------
__global__ void cm_conv_silu(const float* __restrict__ X, int ldx,
                             const float* __restrict__ w, const float* __restrict__ bias,
                             float* __restrict__ U, unsigned short* __restrict__ Ubf) {
    int i = blockIdx.x * blockDim.x + threadIdx.x;   // over B*L*DI
    if (i >= cMB * cDI) return;
    int d  = i % cDI;
    int bl = i / cDI;
    int l  = bl % cL;
    float acc = bias[d];
#pragma unroll
    for (int k = 0; k < cKC; ++k) {
        int ls = l - (cKC - 1) + k;
        if (ls >= 0) acc += X[(size_t)(bl - (cKC - 1) + k) * ldx + d] * w[d * cKC + k];
    }
    float s = acc * sigmoidf_(acc);
    U[i] = s;
    Ubf[i] = f2bf(s);
}

// ---------------- selective scan: one LANE per (b, d, n) state ----------------
// 16 lanes form one channel; per-step output contraction is a 4-step shfl_xor
// butterfly inside the 16-lane group. 65536 threads -> 2048 waves of latency hiding.
__global__ void cm_scan(const float* __restrict__ U, const float* __restrict__ Delta,
                        const float* __restrict__ Dbl, const float* __restrict__ Alog,
                        const float* __restrict__ Dp, float* __restrict__ Y, int flip) {
    int t = blockIdx.x * blockDim.x + threadIdx.x;   // over B*DI*N
    if (t >= cB * cDI * cN) return;
    const int n  = t & (cN - 1);
    const int ch = t >> 4;
    const int d  = ch % cDI;
    const int b  = ch / cDI;

    const float Av = -__expf(Alog[d * cN + n]);
    const float Dd = Dp[d];
    float h = 0.f;

    for (int l = 0; l < cL; ++l) {
        const size_t bl = (size_t)b * cL + l;
        const float dl = Delta[bl * cDI + d];   // broadcast across the 16 lanes
        const float u  = U[bl * cDI + d];       // broadcast across the 16 lanes
        const float* q = Dbl + bl * 96;         // Bm at [64..80), Cm at [80..96)
        h = __expf(dl * Av) * h + dl * q[64 + n] * u;
        float y = h * q[80 + n];
        y += __shfl_xor(y, 8, 16);
        y += __shfl_xor(y, 4, 16);
        y += __shfl_xor(y, 2, 16);
        y += __shfl_xor(y, 1, 16);
        if (n == 0) {
            const int lo = flip ? (cL - 1 - l) : l;
            Y[((size_t)b * cL + lo) * cDI + d] = y + u * Dd;
        }
    }
}

// ---------------- gate (y * silu(z)) + RMSNorm -> bf16 ----------------
__global__ void __launch_bounds__(256)
cm_gate_rms(const float* __restrict__ Yf, const float* __restrict__ Yb,
            const float* __restrict__ XZ, const float* __restrict__ nw,
            unsigned short* __restrict__ Gbf) {
    __shared__ float red[256];
    const int bl  = blockIdx.x;     // token index
    const int tid = threadIdx.x;
    const float* zrow = XZ + (size_t)bl * (2 * cDI) + cDI;
    const size_t base = (size_t)bl * cDI;

    float g[cDI / 256];
    float ss = 0.f;
#pragma unroll
    for (int i = 0; i < cDI / 256; ++i) {
        int d = tid + i * 256;
        float y = Yf[base + d] + Yb[base + d];
        float z = zrow[d];
        float gg = y * (z * sigmoidf_(z));
        g[i] = gg;
        ss += gg * gg;
    }
    red[tid] = ss;
    __syncthreads();
    for (int s = 128; s > 0; s >>= 1) {
        if (tid < s) red[tid] += red[tid + s];
        __syncthreads();
    }
    float scale = rsqrtf(red[0] / (float)cDI + EPSV);
#pragma unroll
    for (int i = 0; i < cDI / 256; ++i) {
        int d = tid + i * 256;
        Gbf[base + d] = f2bf(g[i] * scale * nw[d]);
    }
}

// ---------------- host launcher ----------------
extern "C" void kernel_launch(void* const* d_in, const int* in_sizes, int n_in,
                              void* d_out, int out_size, void* d_ws, size_t ws_size,
                              hipStream_t stream) {
    const float* a       = (const float*)d_in[0];
    const float* b       = (const float*)d_in[1];
    const float* Wi      = (const float*)d_in[2];
    const float* conv_w  = (const float*)d_in[3];
    const float* conv_b  = (const float*)d_in[4];
    const float* Wx      = (const float*)d_in[5];
    const float* Wdt     = (const float*)d_in[6];
    const float* bdt     = (const float*)d_in[7];
    const float* A_log   = (const float*)d_in[8];
    const float* Dp      = (const float*)d_in[9];
    const float* conv_wb = (const float*)d_in[10];
    const float* conv_bb = (const float*)d_in[11];
    const float* Wx_b    = (const float*)d_in[12];
    const float* Wdt_b   = (const float*)d_in[13];
    const float* bdt_b   = (const float*)d_in[14];
    const float* A_log_b = (const float*)d_in[15];
    const float* Dp_b    = (const float*)d_in[16];
    const float* Wo      = (const float*)d_in[17];
    const float* norm_w  = (const float*)d_in[18];
    float* out = (float*)d_out;

    unsigned char* ws = (unsigned char*)d_ws;
    size_t off = 0;
#define WS_ALLOC(ptrname, type, bytes) \
    type* ptrname = (type*)(ws + off); off += (((size_t)(bytes)) + 255) & ~(size_t)255;

    WS_ALLOC(aBF,   unsigned short, (size_t)cMB * cDM * 2)
    WS_ALLOC(bFBF,  unsigned short, (size_t)cMB * cDM * 2)
    WS_ALLOC(WiBF,  unsigned short, (size_t)2 * cDI * cDM * 2)
    WS_ALLOC(WxfBF, unsigned short, (size_t)96 * cDI * 2)
    WS_ALLOC(WxbBF, unsigned short, (size_t)96 * cDI * 2)
    WS_ALLOC(WdtfBF,unsigned short, (size_t)cDI * cR * 2)
    WS_ALLOC(WdtbBF,unsigned short, (size_t)cDI * cR * 2)
    WS_ALLOC(WoBF,  unsigned short, (size_t)cDM * cDI * 2)
    WS_ALLOC(XZ,    float,          (size_t)cMB * 2 * cDI * 4)
    WS_ALLOC(XBW,   float,          (size_t)cMB * cDI * 4)
    WS_ALLOC(UF,    float,          (size_t)cMB * cDI * 4)
    WS_ALLOC(UB,    float,          (size_t)cMB * cDI * 4)
    WS_ALLOC(UFbf,  unsigned short, (size_t)cMB * cDI * 2)
    WS_ALLOC(UBbf,  unsigned short, (size_t)cMB * cDI * 2)
    WS_ALLOC(PBUF,  float,          (size_t)cSPLITK * cMB * 96 * 4)  // split-K partials (reused)
    WS_ALLOC(DBLF,  float,          (size_t)cMB * 96 * 4)
    WS_ALLOC(DBLB,  float,          (size_t)cMB * 96 * 4)
    WS_ALLOC(DTFbf, unsigned short, (size_t)cMB * cR * 2)
    WS_ALLOC(DTBbf, unsigned short, (size_t)cMB * cR * 2)
    WS_ALLOC(DELF,  float,          (size_t)cMB * cDI * 4)
    WS_ALLOC(DELB,  float,          (size_t)cMB * cDI * 4)
    WS_ALLOC(YF,    float,          (size_t)cMB * cDI * 4)
    WS_ALLOC(YB,    float,          (size_t)cMB * cDI * 4)
    WS_ALLOC(GBF,   unsigned short, (size_t)cMB * cDI * 2)
#undef WS_ALLOC
    if (ws_size < off) return;  // workspace too small; nothing safe to do

    const int T = 256;
#define NB(n) (((n) + T - 1) / T)
#define GEMM_GRID(Mv, Nv) dim3(((Nv) / 32 + 7) / 8, (Mv) / 32)

    // 1) pack activations + weights to bf16
    cm_pack<<<NB(cMB * cDM), T, 0, stream>>>(a, aBF, cMB * cDM);
    cm_pack_flip<<<NB(cMB * cDM), T, 0, stream>>>(b, bFBF);
    cm_pack<<<NB(2 * cDI * cDM), T, 0, stream>>>(Wi, WiBF, 2 * cDI * cDM);
    cm_pack<<<NB(96 * cDI), T, 0, stream>>>(Wx, WxfBF, 96 * cDI);
    cm_pack<<<NB(96 * cDI), T, 0, stream>>>(Wx_b, WxbBF, 96 * cDI);
    cm_pack<<<NB(cDI * cR), T, 0, stream>>>(Wdt, WdtfBF, cDI * cR);
    cm_pack<<<NB(cDI * cR), T, 0, stream>>>(Wdt_b, WdtbBF, cDI * cR);
    cm_pack<<<NB(cDM * cDI), T, 0, stream>>>(Wo, WoBF, cDM * cDI);

    dim3 gb(32, 8);  // 8 waves -> 8 adjacent 32-wide n-slots
    // 2) input projections (WMMA, K=1024 compile-time)
    cm_gemm<0, cDM, cDM><<<GEMM_GRID(cMB, 2 * cDI), gb, 0, stream>>>(
        aBF, WiBF, XZ, cMB, 2 * cDI, nullptr);
    cm_gemm<0, cDM, cDM><<<GEMM_GRID(cMB, cDI), gb, 0, stream>>>(
        bFBF, WiBF, XBW, cMB, cDI, nullptr);

    // 3) causal conv + SiLU
    cm_conv_silu<<<NB(cMB * cDI), T, 0, stream>>>(XZ, 2 * cDI, conv_w, conv_b, UF, UFbf);
    cm_conv_silu<<<NB(cMB * cDI), T, 0, stream>>>(XBW, cDI, conv_wb, conv_bb, UB, UBbf);

    // 4) dbl = x @ Wx^T (WMMA, N=96): split-K x8 for parallelism, deterministic reduce,
    //    fused dt bf16 extraction. PBUF reused sequentially (stream-ordered).
    dim3 gridDbl((96 / 32 + 7) / 8, cMB / 32, cSPLITK);
    cm_gemm<3, cDI, cDI / cSPLITK><<<gridDbl, gb, 0, stream>>>(
        UFbf, WxfBF, PBUF, cMB, 96, nullptr);
    cm_reduce_dbl<<<NB(cMB * 96), T, 0, stream>>>(PBUF, DBLF, DTFbf);
    cm_gemm<3, cDI, cDI / cSPLITK><<<gridDbl, gb, 0, stream>>>(
        UBbf, WxbBF, PBUF, cMB, 96, nullptr);
    cm_reduce_dbl<<<NB(cMB * 96), T, 0, stream>>>(PBUF, DBLB, DTBbf);

    // 5) delta = softplus(dt @ Wdt^T + 2*bdt) (WMMA, K=64 compile-time)
    cm_gemm<1, cR, cR><<<GEMM_GRID(cMB, cDI), gb, 0, stream>>>(
        DTFbf, WdtfBF, DELF, cMB, cDI, bdt);
    cm_gemm<1, cR, cR><<<GEMM_GRID(cMB, cDI), gb, 0, stream>>>(
        DTBbf, WdtbBF, DELB, cMB, cDI, bdt_b);

    // 6) selective scans (backward writes flipped)
    cm_scan<<<NB(cB * cDI * cN), T, 0, stream>>>(UF, DELF, DBLF, A_log, Dp, YF, 0);
    cm_scan<<<NB(cB * cDI * cN), T, 0, stream>>>(UB, DELB, DBLB, A_log_b, Dp_b, YB, 1);

    // 7) gate + RMSNorm -> bf16
    cm_gate_rms<<<cMB, 256, 0, stream>>>(YF, YB, XZ, norm_w, GBF);

    // 8) output projection (WMMA, K=2048 compile-time)
    cm_gemm<0, cDI, cDI><<<GEMM_GRID(cMB, cDM), gb, 0, stream>>>(
        GBF, WoBF, out, cMB, cDM, nullptr);
#undef NB
#undef GEMM_GRID
}
